// Model_11613591568858
// MI455X (gfx1250) — compile-verified, matrix operand order checked by reading, not run on hardware
//
#include <hip/hip_runtime.h>

// ---------------------------------------------------------------------------
// MI455X (gfx1250) implementation.
//
// Roofline: ~50 GFLOP total, ~2.5MB of weights (L2-resident), sequential
// T=50 scan => latency-bound recurrent GEMMs. Strategy: bf16 WMMA
// (v_wmma_f32_16x16x32_bf16) with f32 accumulation; one block per 32 batch
// rows runs the whole scan with state in LDS/registers; weights pre-packed
// into WMMA B-fragment order so every lane loads its fragment with two
// b128 loads. DNC memory (512x50x64 f32 = 6.5MB) lives in workspace / L2.
//
// Round-1 fix: the z-GEMM is now a single non-unrolled K loop sharing
// A fragments across n-tiles (2 A-frags + 4 B-frags + 64 accum VGPRs live)
// to eliminate the scratch spills seen in the previous build.
// ---------------------------------------------------------------------------

typedef __attribute__((ext_vector_type(16))) __bf16 bf16x16;
typedef __attribute__((ext_vector_type(8)))  float  f32x8;

#define B_    512
#define T_    50
#define U_    256
#define CS_   64
#define R_    4
#define M_    50
#define CLS_  5

// A-panel column layout (per batch row, 672 bf16):
//   [0..255]   h
//   [256..511] read (r*64+c)
//   [512..644] x   (128 conv feats + 5 one-hot)
//   [645..671] zero pad
#define KZ   672   // 21 k-blocks of 32
#define NZ   1024  // 64 n-tiles
#define KP   256   // 8 k-blocks
#define NP   320   // 20 n-tiles
#define KO   512   // 16 k-blocks
#define NO   256   // 16 n-tiles
#define XSTR 144   // x_all row stride (bf16)

__device__ inline __bf16 f2bf(float f) {
    union { float f; unsigned u; } in; in.f = f;
    unsigned u = in.u;
    unsigned r = u + 0x7FFFu + ((u >> 16) & 1u);
    union { unsigned short s; __bf16 b; } out;
    out.s = (unsigned short)(r >> 16);
    return out.b;
}

__device__ inline float sigm(float x) { return 1.0f / (1.0f + __expf(-x)); }

__device__ inline bf16x16 mk16(float4 a, float4 b) {
    union { float4 f; __bf16 h[8]; } ua, ub;
    ua.f = a; ub.f = b;
    bf16x16 r;
#pragma unroll
    for (int i = 0; i < 8; ++i) { r[i] = ua.h[i]; r[i + 8] = ub.h[i]; }
    return r;
}

// A fragment (16-bit A 16x32 layout): lane (g=lane>>4, m=lane&15) reads two
// contiguous 8-element chunks at cols kb*32+8g and kb*32+8g+16 of LDS panel.
__device__ inline bf16x16 load_afrag(const __bf16* A, int row0, int kb, int g, int m) {
    const __bf16* arow = A + (size_t)(row0 + m) * KZ + 8 * g + kb * 32;
    const float4* ap = reinterpret_cast<const float4*>(arow);
    return mk16(ap[0], ap[2]);
}

// B fragment from fragment-packed global weights:
//   Bp[((kb*NT + nt)*32 + lane)*16 + c*8 + j]
__device__ inline bf16x16 load_bfrag(const __bf16* Bp, int NT, int kb, int nt, int lane) {
    const float4* bp = reinterpret_cast<const float4*>(
        Bp + (((size_t)kb * NT + nt) * 32 + lane) * 16);
    return mk16(bp[0], bp[1]);
}

// One 16x16 output tile with a single live accumulator (used for the small
// param / out GEMMs where full K unroll does not create register pressure).
template<int NKB, int NT>
__device__ inline f32x8 wmma_tiles(const __bf16* A, int row0,
                                   const __bf16* Bp, int nt, int lane) {
    const int g = lane >> 4, m = lane & 15;
    f32x8 acc;
#pragma unroll
    for (int i = 0; i < 8; ++i) acc[i] = 0.0f;
#pragma unroll
    for (int kb = 0; kb < NKB; ++kb) {
        bf16x16 av = load_afrag(A, row0, kb, g, m);
        bf16x16 bv = load_bfrag(Bp, NT, kb, nt, lane);
        acc = __builtin_amdgcn_wmma_f32_16x16x32_bf16(false, av, false, bv,
                                                      (short)0, acc, false, false);
    }
    return acc;
}

// ---------------------------------------------------------------------------
// Kernel 1: pack weights (f32 -> bf16) into WMMA B-fragment order.
//   Wz rows: [0..255]=lstm_r, [256..511]=lstm_k[133..388], [512..644]=lstm_k[0..132], rest 0.
// ---------------------------------------------------------------------------
__global__ __launch_bounds__(256) void pack_kernel(
    const float* __restrict__ lstm_k, const float* __restrict__ lstm_r,
    const float* __restrict__ param_w, const float* __restrict__ out_w,
    __bf16* __restrict__ wz, __bf16* __restrict__ wp, __bf16* __restrict__ wo)
{
    size_t idx = (size_t)blockIdx.x * 256 + threadIdx.x;
    const size_t NZc = 21ull * 64 * 512;   // 688128
    const size_t NPc = 8ull  * 20 * 512;   // 81920
    const size_t NOc = 16ull * 16 * 512;   // 131072
    if (idx < NZc) {
        size_t e = idx;
        int j = e & 7, c = (e >> 3) & 1, ln = (e >> 4) & 31;
        size_t r2 = e >> 9;
        int nt = (int)(r2 % 64), kb = (int)(r2 / 64);
        int K = kb * 32 + 8 * (ln >> 4) + 16 * c + j;
        int n = nt * 16 + (ln & 15);
        float v = 0.0f;
        if      (K < 256) v = lstm_r[(size_t)K * 1024 + n];
        else if (K < 512) v = lstm_k[(size_t)(K - 256 + 133) * 1024 + n];
        else if (K < 645) v = lstm_k[(size_t)(K - 512) * 1024 + n];
        wz[idx] = f2bf(v);
    } else if (idx < NZc + NPc) {
        size_t e = idx - NZc;
        int j = e & 7, c = (e >> 3) & 1, ln = (e >> 4) & 31;
        size_t r2 = e >> 9;
        int nt = (int)(r2 % 20), kb = (int)(r2 / 20);
        int K = kb * 32 + 8 * (ln >> 4) + 16 * c + j;
        int n = nt * 16 + (ln & 15);
        wp[e] = f2bf(param_w[(size_t)K * 320 + n]);
    } else if (idx < NZc + NPc + NOc) {
        size_t e = idx - NZc - NPc;
        int j = e & 7, c = (e >> 3) & 1, ln = (e >> 4) & 31;
        size_t r2 = e >> 9;
        int nt = (int)(r2 % 16), kb = (int)(r2 / 16);
        int K = kb * 32 + 8 * (ln >> 4) + 16 * c + j;
        int n = nt * 16 + (ln & 15);
        wo[e] = f2bf(out_w[(size_t)K * 256 + n]);
    }
}

// ---------------------------------------------------------------------------
// Kernel 2: conv front-end. One block per (b,t) image. BN folded to
// scale/bias, /255 folded into the input load. Writes x_all[t][b][0..132]
// (features + one-hot) as bf16, cols 133..143 zero.
// ---------------------------------------------------------------------------
__global__ __launch_bounds__(128) void conv_kernel(
    const float* __restrict__ images, const int* __restrict__ prev,
    const float* __restrict__ w1, const float* __restrict__ g1, const float* __restrict__ b1,
    const float* __restrict__ m1, const float* __restrict__ v1,
    const float* __restrict__ w2, const float* __restrict__ g2, const float* __restrict__ b2,
    const float* __restrict__ m2, const float* __restrict__ v2,
    const float* __restrict__ w3, const float* __restrict__ g3, const float* __restrict__ b3,
    const float* __restrict__ m3, const float* __restrict__ v3,
    __bf16* __restrict__ xall)
{
    __shared__ float img[28 * 28];
    __shared__ float a1[13 * 13 * 8];
    __shared__ float a2[6 * 6 * 16];
    __shared__ float s1[8], o1[8], s2[16], o2[16], s3[32], o3[32];
    const int tid = threadIdx.x;
    const int im = blockIdx.x;            // b*T + t
    const int b = im / T_, t = im % T_;

    if (tid < 8)       { float s = g1[tid] * rsqrtf(v1[tid] + 1e-3f); s1[tid] = s; o1[tid] = b1[tid] - m1[tid] * s; }
    else if (tid < 24) { int c = tid - 8;  float s = g2[c] * rsqrtf(v2[c] + 1e-3f); s2[c] = s; o2[c] = b2[c] - m2[c] * s; }
    else if (tid < 56) { int c = tid - 24; float s = g3[c] * rsqrtf(v3[c] + 1e-3f); s3[c] = s; o3[c] = b3[c] - m3[c] * s; }
    for (int i = tid; i < 784; i += 128)
        img[i] = images[(size_t)im * 784 + i] * (1.0f / 255.0f);
    __syncthreads();

    for (int i = tid; i < 13 * 13 * 8; i += 128) {
        int oc = i & 7, xi = (i >> 3) % 13, yi = (i >> 3) / 13;
        float s = 0.0f;
#pragma unroll
        for (int ky = 0; ky < 3; ++ky)
#pragma unroll
            for (int kx = 0; kx < 3; ++kx)
                s += img[(2 * yi + ky) * 28 + 2 * xi + kx] * w1[(ky * 3 + kx) * 8 + oc];
        a1[i] = fmaxf(s * s1[oc] + o1[oc], 0.0f);
    }
    __syncthreads();

    for (int i = tid; i < 6 * 6 * 16; i += 128) {
        int oc = i & 15, xi = (i >> 4) % 6, yi = (i >> 4) / 6;
        float s = 0.0f;
#pragma unroll
        for (int ky = 0; ky < 3; ++ky)
#pragma unroll
            for (int kx = 0; kx < 3; ++kx)
                for (int ic = 0; ic < 8; ++ic)
                    s += a1[((2 * yi + ky) * 13 + 2 * xi + kx) * 8 + ic] *
                         w2[((ky * 3 + kx) * 8 + ic) * 16 + oc];
        a2[i] = fmaxf(s * s2[oc] + o2[oc], 0.0f);
    }
    __syncthreads();

    __bf16* xrow = xall + ((size_t)t * B_ + b) * XSTR;
    if (tid < 128) {
        int i = tid;
        int oc = i & 31, xi = (i >> 5) & 1, yi = i >> 6;
        float s = 0.0f;
#pragma unroll
        for (int ky = 0; ky < 3; ++ky)
#pragma unroll
            for (int kx = 0; kx < 3; ++kx)
                for (int ic = 0; ic < 16; ++ic)
                    s += a2[((2 * yi + ky) * 6 + 2 * xi + kx) * 16 + ic] *
                         w3[((ky * 3 + kx) * 16 + ic) * 32 + oc];
        xrow[i] = f2bf(fmaxf(s * s3[oc] + o3[oc], 0.0f));
    }
    if (tid < 16) {
        int col = 128 + tid;
        float v = 0.0f;
        if (col < 133) v = (prev[b * T_ + t] == tid) ? 1.0f : 0.0f;
        xrow[col] = f2bf(v);
    }
}

// ---------------------------------------------------------------------------
// Kernel 3: the recurrent scan. One block = 32 batch rows, 512 threads
// (16 wave32s). Everything for those rows stays inside the block for all
// 50 steps; DNC memory in global workspace (L2-resident).
// ---------------------------------------------------------------------------
__global__ __launch_bounds__(512) void rec_kernel(
    const __bf16* __restrict__ xall,
    const __bf16* __restrict__ Wz, const __bf16* __restrict__ Wp, const __bf16* __restrict__ Wo,
    const float* __restrict__ lstmb, const float* __restrict__ pb,
    const float* __restrict__ ob, const float* __restrict__ hw, const float* __restrict__ hb,
    float* __restrict__ memg, float* __restrict__ outp)
{
    __shared__ __align__(16) __bf16 Az[32 * KZ];            // 43008 B
    __shared__ float fbuf[32 * NP + 32 * NO];               // 73728 B: inter | outb
    __shared__ float cbuf[32 * U_];                          // 32768 B
    __shared__ float invn[32 * M_];                          // 6400 B
    float* inter = fbuf;
    float* outb  = fbuf + 32 * NP;

    const int tid  = threadIdx.x;
    const int lane = tid & 31, w = tid >> 5;
    const int g    = lane >> 4, nn = lane & 15;
    const int gb0  = blockIdx.x * 32;
    float* mymem = memg + (size_t)gb0 * (M_ * CS_);

    // ---- init: zero A-panel (h=read=0, pad=0), cell state, DNC memory ----
    for (int i = tid; i < 32 * KZ; i += 512) Az[i] = f2bf(0.0f);
    for (int i = tid; i < 32 * U_; i += 512) cbuf[i] = 0.0f;
    for (int i = tid; i < 32 * M_ * CS_; i += 512) mymem[i] = 0.0f;
    __syncthreads();

#pragma clang loop unroll(disable)
    for (int t = 0; t < T_; ++t) {
        // ---- load x_t into A-panel cols [512..644] ----
        for (int i = tid; i < 32 * 133; i += 512) {
            int b = i / 133, col = i % 133;
            Az[b * KZ + 512 + col] = xall[((size_t)t * B_ + gb0 + b) * XSTR + col];
        }
        __syncthreads();

        // ---- z GEMM: 32x1024, K=672. Wave w owns n-tiles {w,w+16,w+32,w+48}
        // for both m-tiles, so each lane holds i/f/g/o for its (b,u) in regs.
        // Single rolled K loop sharing A/B fragments across tiles: live state
        // is 64 accum + 16 A + 32 B VGPRs -> no scratch spills.
        f32x8 zacc[2][4];
#pragma unroll
        for (int mt = 0; mt < 2; ++mt)
#pragma unroll
            for (int q = 0; q < 4; ++q)
#pragma unroll
                for (int i = 0; i < 8; ++i) zacc[mt][q][i] = 0.0f;
#pragma clang loop unroll(disable)
        for (int kb = 0; kb < 21; ++kb) {
            bf16x16 a0 = load_afrag(Az, 0,  kb, g, nn);
            bf16x16 a1 = load_afrag(Az, 16, kb, g, nn);
#pragma unroll
            for (int q = 0; q < 4; ++q) {
                bf16x16 bv = load_bfrag(Wz, 64, kb, w + 16 * q, lane);
                zacc[0][q] = __builtin_amdgcn_wmma_f32_16x16x32_bf16(
                    false, a0, false, bv, (short)0, zacc[0][q], false, false);
                zacc[1][q] = __builtin_amdgcn_wmma_f32_16x16x32_bf16(
                    false, a1, false, bv, (short)0, zacc[1][q], false, false);
            }
        }
        __syncthreads();   // everyone done reading Az before h is rewritten

        // ---- LSTM gates entirely in registers ----
        {
            const int u = w * 16 + nn;
            float bi  = lstmb[u],       bff = lstmb[u + 256];
            float bgg = lstmb[u + 512], boo = lstmb[u + 768];
#pragma unroll
            for (int mt = 0; mt < 2; ++mt) {
#pragma unroll
                for (int v = 0; v < 8; ++v) {
                    int b = mt * 16 + v + 8 * g;
                    float zi = zacc[mt][0][v] + bi;
                    float zf = zacc[mt][1][v] + bff;
                    float zg = zacc[mt][2][v] + bgg;
                    float zo = zacc[mt][3][v] + boo;
                    float co = cbuf[b * U_ + u];
                    float cn = sigm(zf) * co + sigm(zi) * tanhf(zg);
                    float hn = sigm(zo) * tanhf(cn);
                    cbuf[b * U_ + u] = cn;
                    Az[b * KZ + u] = f2bf(hn);   // h region
                }
            }
        }
        __syncthreads();

        // ---- param GEMM: inter = tanh(h @ Wp + pb), 32x320, K=256 ----
        for (int ti = w; ti < 40; ti += 16) {
            int mt = ti / 20, nt = ti % 20;
            f32x8 acc = wmma_tiles<8, 20>(Az, mt * 16, Wp, nt, lane);
            int col = nt * 16 + nn;
            float bv = pb[col];
#pragma unroll
            for (int v = 0; v < 8; ++v) {
                int row = mt * 16 + v + 8 * g;
                inter[row * NP + col] = tanhf(acc[v] + bv);
            }
        }
        __syncthreads();

        // ---- attention over memory (order-invariant => no ring rotation) ----
        for (int i = tid; i < 32 * M_; i += 512) {
            int b = i / M_, m = i % M_;
            const float* mp = mymem + (b * M_ + m) * CS_;
            float s = 0.0f;
            for (int c2 = 0; c2 < CS_; ++c2) s += mp[c2] * mp[c2];
            invn[i] = rsqrtf(fmaxf(s, 1e-12f));
        }
        __syncthreads();
        if (tid < 128) {
            int b = tid >> 2, r = tid & 3;
            const float* key = inter + b * NP + CS_ + r * CS_;
            float ks = 0.0f;
            for (int c2 = 0; c2 < CS_; ++c2) ks += key[c2] * key[c2];
            float kin = rsqrtf(fmaxf(ks, 1e-12f));
            const float* mp = mymem + b * (M_ * CS_);
            float att[M_], amax = -1e30f;
#pragma unroll
            for (int m = 0; m < M_; ++m) {
                float d = 0.0f;
                for (int c2 = 0; c2 < CS_; ++c2) d += key[c2] * mp[m * CS_ + c2];
                float a = d * kin * invn[b * M_ + m];
                att[m] = a; amax = fmaxf(amax, a);
            }
            float asum = 0.0f;
#pragma unroll
            for (int m = 0; m < M_; ++m) { att[m] = __expf(att[m] - amax); asum += att[m]; }
            float ainv = 1.0f / asum;
            for (int c2 = 0; c2 < CS_; ++c2) {
                float acc = 0.0f;
#pragma unroll
                for (int m = 0; m < M_; ++m) acc += att[m] * mp[m * CS_ + c2];
                Az[b * KZ + 256 + r * CS_ + c2] = f2bf(acc * ainv);  // read region
            }
        }
        __syncthreads();

        // ---- memory write (T==M: physical row t is the dropped zero row) ----
        {
            int wr = t % M_;
            for (int i = tid; i < 32 * CS_; i += 512) {
                int b = i >> 6, c2 = i & 63;
                mymem[(b * M_ + wr) * CS_ + c2] = inter[b * NP + c2];
            }
        }

        // ---- out GEMM: tanh([h|read] @ Wo + ob), 32x256, K=512 ----
#pragma unroll
        for (int ti = w * 2; ti < w * 2 + 2; ++ti) {
            int mt = ti >> 4, nt = ti & 15;
            f32x8 acc = wmma_tiles<16, 16>(Az, mt * 16, Wo, nt, lane);
            int col = nt * 16 + nn;
            float bv = ob[col];
#pragma unroll
            for (int v = 0; v < 8; ++v) {
                int row = mt * 16 + v + 8 * g;
                outb[row * NO + col] = tanhf(acc[v] + bv);
            }
        }
        __syncthreads();

        // ---- classifier head + softmax (tiny: VALU) ----
        if (tid < 32) {
            int b = tid;
            float lg[CLS_];
#pragma unroll
            for (int cl = 0; cl < CLS_; ++cl) lg[cl] = hb[cl];
            for (int u = 0; u < U_; ++u) {
                float hv = outb[b * NO + u];
#pragma unroll
                for (int cl = 0; cl < CLS_; ++cl) lg[cl] += hv * hw[u * CLS_ + cl];
            }
            float mx = lg[0];
#pragma unroll
            for (int cl = 1; cl < CLS_; ++cl) mx = fmaxf(mx, lg[cl]);
            float sm = 0.0f;
#pragma unroll
            for (int cl = 0; cl < CLS_; ++cl) { lg[cl] = __expf(lg[cl] - mx); sm += lg[cl]; }
            float si = 1.0f / sm;
#pragma unroll
            for (int cl = 0; cl < CLS_; ++cl)
                outp[((size_t)(gb0 + b) * T_ + t) * CLS_ + cl] = lg[cl] * si;
        }
        __syncthreads();
    }
}

// ---------------------------------------------------------------------------
// Workspace layout
// ---------------------------------------------------------------------------
static constexpr size_t SZ_XALL = (size_t)T_ * B_ * XSTR * 2;     // 7,372,800
static constexpr size_t OFF_WZ  = SZ_XALL;
static constexpr size_t SZ_WZ   = 21ull * 64 * 512 * 2;           // 1,376,256
static constexpr size_t OFF_WP  = OFF_WZ + SZ_WZ;
static constexpr size_t SZ_WP   = 8ull * 20 * 512 * 2;            // 163,840
static constexpr size_t OFF_WO  = OFF_WP + SZ_WP;
static constexpr size_t SZ_WO   = 16ull * 16 * 512 * 2;           // 262,144
static constexpr size_t OFF_MEM = OFF_WO + SZ_WO;                 // + 6,553,600 f32

extern "C" void kernel_launch(void* const* d_in, const int* in_sizes, int n_in,
                              void* d_out, int out_size, void* d_ws, size_t ws_size,
                              hipStream_t stream) {
    (void)in_sizes; (void)n_in; (void)out_size; (void)ws_size;
    const float* images  = (const float*)d_in[0];
    const int*   prev    = (const int*)  d_in[1];
    const float* conv1_w = (const float*)d_in[2];
    const float* bn1_g   = (const float*)d_in[3];
    const float* bn1_b   = (const float*)d_in[4];
    const float* bn1_m   = (const float*)d_in[5];
    const float* bn1_v   = (const float*)d_in[6];
    const float* conv2_w = (const float*)d_in[7];
    const float* bn2_g   = (const float*)d_in[8];
    const float* bn2_b   = (const float*)d_in[9];
    const float* bn2_m   = (const float*)d_in[10];
    const float* bn2_v   = (const float*)d_in[11];
    const float* conv3_w = (const float*)d_in[12];
    const float* bn3_g   = (const float*)d_in[13];
    const float* bn3_b   = (const float*)d_in[14];
    const float* bn3_m   = (const float*)d_in[15];
    const float* bn3_v   = (const float*)d_in[16];
    const float* lstm_k  = (const float*)d_in[17];
    const float* lstm_r  = (const float*)d_in[18];
    const float* lstm_b  = (const float*)d_in[19];
    const float* param_w = (const float*)d_in[20];
    const float* param_b = (const float*)d_in[21];
    const float* out_w   = (const float*)d_in[22];
    const float* out_b   = (const float*)d_in[23];
    const float* head_w  = (const float*)d_in[24];
    const float* head_b  = (const float*)d_in[25];

    char* ws = (char*)d_ws;
    __bf16* xall = (__bf16*)(ws);
    __bf16* wz   = (__bf16*)(ws + OFF_WZ);
    __bf16* wp   = (__bf16*)(ws + OFF_WP);
    __bf16* wo   = (__bf16*)(ws + OFF_WO);
    float*  memg = (float*) (ws + OFF_MEM);
    float*  outp = (float*)d_out;

    // 1) pack weights into WMMA fragment order (688128+81920+131072 elems)
    pack_kernel<<<3520, 256, 0, stream>>>(lstm_k, lstm_r, param_w, out_w, wz, wp, wo);

    // 2) conv front-end: one block per image
    conv_kernel<<<B_ * T_, 128, 0, stream>>>(images, prev,
        conv1_w, bn1_g, bn1_b, bn1_m, bn1_v,
        conv2_w, bn2_g, bn2_b, bn2_m, bn2_v,
        conv3_w, bn3_g, bn3_b, bn3_m, bn3_v, xall);

    // 3) recurrent scan: 16 blocks x 512 threads, whole T=50 loop per block
    rec_kernel<<<B_ / 32, 512, 0, stream>>>(xall, wz, wp, wo,
        lstm_b, param_b, out_b, head_w, head_b, memg, outp);
}
